// RecCore_76613626626072
// MI455X (gfx1250) — compile-verified, matrix operand order checked by reading, not run on hardware
//
#include <hip/hip_runtime.h>

// ---------------- problem constants ----------------
#define BB 8192      // batch rows
#define HH 2048      // hidden size
#define KK 2048      // reduction dim (D == H == 2048)
#define H3 6144      // 3*H
// ---------------- tiling ----------------
#define BM 128       // M rows per block
#define BN 64        // N cols per gate per block
#define KB 32        // K per stage (bf16 WMMA K)
#define NK (KK / KB) // 64 k-steps
#define NG 3         // gates (r, z, n)
#define STR (KB + 8) // padded LDS row stride (80B -> spreads banks, 16B aligned)

typedef __attribute__((ext_vector_type(16))) __bf16 v16bf;
typedef __attribute__((ext_vector_type(8)))  __bf16 v8bf;
typedef __attribute__((ext_vector_type(8)))  float  v8f;
typedef __attribute__((ext_vector_type(4)))  int    v4i;

// ------------- async global->LDS path (CDNA5), guarded -------------
#if defined(__HIP_DEVICE_COMPILE__) && __has_builtin(__builtin_amdgcn_global_load_async_to_lds_b128)
#define USE_ASYNC 1
#endif

#if defined(USE_ASYNC)
// Builtin signature (from clang diagnostic): param0 = v4i in AS1 (global),
// param1 = v4i in AS3 (LDS), then imm offset, imm cpol.
typedef __attribute__((address_space(1))) v4i gv4i;
typedef __attribute__((address_space(3))) v4i lv4i;
#define CP16(l, g)                                                     \
  __builtin_amdgcn_global_load_async_to_lds_b128(                      \
      (gv4i*)(unsigned long long)(g),                                  \
      (lv4i*)(unsigned int)(unsigned long long)(l), 0, 0)
#if __has_builtin(__builtin_amdgcn_s_wait_asynccnt)
#define WAIT_ASYNC() __builtin_amdgcn_s_wait_asynccnt(0)
#else
#define WAIT_ASYNC() asm volatile("s_wait_asynccnt 0" ::: "memory")
#endif
#else
#define CP16(l, g) (*(v8bf*)(l) = *(const v8bf*)(g))
#define WAIT_ASYNC() ((void)0)
#endif

// ------------- fast activations (v_exp_f32 / v_rcp_f32) -------------
__device__ __forceinline__ float sigm_f(float x) {
  return __builtin_amdgcn_rcpf(1.f + __expf(-x));
}
__device__ __forceinline__ float tanh_f(float x) {
  // tanh(x) = 1 - 2/(exp(2x)+1)
  return 1.f - 2.f * __builtin_amdgcn_rcpf(1.f + __expf(2.f * x));
}

// ------------- one LDS stage: X tile (BM x KB) + W tiles (NG*BN x KB) -------------
__device__ __forceinline__ void stage_load(const __bf16* __restrict__ A,
                                           const __bf16* __restrict__ W,
                                           __bf16* xs, __bf16* wsh,
                                           int mB, int nB, int k0, int tid) {
  // X tile: 128 rows * 32 bf16 = 512 x 16B chunks, 256 threads -> 2 each
#pragma unroll
  for (int i = 0; i < 2; ++i) {
    int c   = tid + i * 256;
    int row = c >> 2;
    int off = (c & 3) * 8;
    const __bf16* g = A + (size_t)(mB * BM + row) * KK + k0 + off;
    __bf16*       l = xs + row * STR + off;
    CP16(l, g);
  }
  // W tiles: per gate 64 rows * 32 bf16 = 256 x 16B chunks -> 1 per thread per gate
#pragma unroll
  for (int g = 0; g < NG; ++g) {
    int row = tid >> 2;
    int off = (tid & 3) * 8;
    const __bf16* gp = W + (size_t)(g * HH + nB * BN + row) * KK + k0 + off;
    __bf16*       lp = wsh + (g * BN + row) * STR + off;
    CP16(lp, gp);
  }
}

// ------------- shared GEMM core: acc[g][mt][nt] += Atile x Wtile(g) -------------
__device__ __forceinline__ void gemm3_core(const __bf16* __restrict__ A,
                                           const __bf16* __restrict__ W,
                                           v8f acc[NG][2][2]) {
  __shared__ __bf16 xs [2][BM * STR];       // 2 x 10 KB
  __shared__ __bf16 wsh[2][NG * BN * STR];  // 2 x 15 KB

  const int tid  = threadIdx.x;
  const int mB   = blockIdx.x, nB = blockIdx.y;
  const int lane = tid & 31;
  const int wid  = tid >> 5;          // 8 waves
  const int wm   = wid >> 1;          // 0..3  -> 32-row strip
  const int wn   = wid & 1;           // 0..1  -> 32-col strip
  const int lh   = lane & 15;
  const int lsel = lane >> 4;

  const v8f z = {0.f, 0.f, 0.f, 0.f, 0.f, 0.f, 0.f, 0.f};
#pragma unroll
  for (int g = 0; g < NG; ++g)
#pragma unroll
    for (int mt = 0; mt < 2; ++mt)
#pragma unroll
      for (int nt = 0; nt < 2; ++nt) acc[g][mt][nt] = z;

  stage_load(A, W, xs[0], wsh[0], mB, nB, 0, tid);
  WAIT_ASYNC();
  __syncthreads();

  for (int ks = 0; ks < NK; ++ks) {
    const int buf = ks & 1;
    if (ks + 1 < NK)
      stage_load(A, W, xs[buf ^ 1], wsh[buf ^ 1], mB, nB, (ks + 1) * KB, tid);

    v16bf afrag[2], bfrag[NG][2];
#pragma unroll
    for (int mt = 0; mt < 2; ++mt)
      afrag[mt] =
          *(const v16bf*)&xs[buf][(wm * 32 + mt * 16 + lh) * STR + lsel * 16];
#pragma unroll
    for (int g = 0; g < NG; ++g)
#pragma unroll
      for (int nt = 0; nt < 2; ++nt)
        bfrag[g][nt] = *(const v16bf*)&wsh[buf][(g * BN + wn * 32 + nt * 16 + lh) * STR +
                                               lsel * 16];

#pragma unroll
    for (int g = 0; g < NG; ++g)
#pragma unroll
      for (int mt = 0; mt < 2; ++mt)
#pragma unroll
        for (int nt = 0; nt < 2; ++nt)
          acc[g][mt][nt] = __builtin_amdgcn_wmma_f32_16x16x32_bf16(
              false, afrag[mt], false, bfrag[g][nt], (short)0, acc[g][mt][nt],
              false, false);

    WAIT_ASYNC();
    __syncthreads();
  }
}

// ------------- kernel 1: gi = enc @ w_ihT + b_ih  (stored bf16) -------------
__global__ __launch_bounds__(256) void gi_gemm_kernel(
    const __bf16* __restrict__ enc, const __bf16* __restrict__ wih,
    const float* __restrict__ bih, __bf16* __restrict__ gi) {
  v8f acc[NG][2][2];
  gemm3_core(enc, wih, acc);

  const int tid = threadIdx.x, lane = tid & 31, wid = tid >> 5;
  const int wm = wid >> 1, wn = wid & 1;
  const int lh = lane & 15, lsel = lane >> 4;
  const int mB = blockIdx.x, nB = blockIdx.y;

#pragma unroll
  for (int g = 0; g < NG; ++g)
#pragma unroll
    for (int mt = 0; mt < 2; ++mt)
#pragma unroll
      for (int nt = 0; nt < 2; ++nt) {
        const int j  = nB * BN + wn * 32 + nt * 16 + lh;
        const int i0 = mB * BM + wm * 32 + mt * 16 + lsel * 8;
        const float bias = bih[g * HH + j];
#pragma unroll
        for (int r = 0; r < 8; ++r)
          gi[(size_t)(i0 + r) * H3 + g * HH + j] =
              (__bf16)(acc[g][mt][nt][r] + bias);
      }
}

// ------------- kernel 2: fused recurrence step -------------
// gh = h @ w_hhT + b_hh, then GRU pointwise, h stays bf16 (ping-pong)
__global__ __launch_bounds__(256) void gru_step_kernel(
    const __bf16* __restrict__ hin, const __bf16* __restrict__ whh,
    const __bf16* __restrict__ gi, const float* __restrict__ bhh,
    __bf16* __restrict__ hout, float* __restrict__ fout) {
  v8f acc[NG][2][2];
  gemm3_core(hin, whh, acc);

  const int tid = threadIdx.x, lane = tid & 31, wid = tid >> 5;
  const int wm = wid >> 1, wn = wid & 1;
  const int lh = lane & 15, lsel = lane >> 4;
  const int mB = blockIdx.x, nB = blockIdx.y;

#pragma unroll
  for (int mt = 0; mt < 2; ++mt)
#pragma unroll
    for (int nt = 0; nt < 2; ++nt) {
      const int j  = nB * BN + wn * 32 + nt * 16 + lh;
      const int i0 = mB * BM + wm * 32 + mt * 16 + lsel * 8;
      const float br = bhh[j];
      const float bz = bhh[HH + j];
      const float bn = bhh[2 * HH + j];
#pragma unroll
      for (int r = 0; r < 8; ++r) {
        const int    i     = i0 + r;
        const size_t gbase = (size_t)i * H3 + j;
        const float  ghr = acc[0][mt][nt][r] + br;
        const float  ghz = acc[1][mt][nt][r] + bz;
        const float  ghn = acc[2][mt][nt][r] + bn;
        const float  rg  = sigm_f((float)gi[gbase] + ghr);
        const float  zg  = sigm_f((float)gi[gbase + HH] + ghz);
        const float  ng  = tanh_f((float)gi[gbase + 2 * HH] + rg * ghn);
        const float  hp  = (float)hin[(size_t)i * HH + j];
        const float  hv  = (1.f - zg) * ng + zg * hp;
        hout[(size_t)i * HH + j] = (__bf16)hv;
        if (fout) fout[(size_t)i * HH + j] = hv;
      }
    }
}

// ------------- helpers -------------
__global__ void cvt_f32_bf16_kernel(const float* __restrict__ in,
                                    __bf16* __restrict__ out, long n) {
  long i = (long)blockIdx.x * blockDim.x + threadIdx.x;
  const long s = (long)gridDim.x * blockDim.x;
  for (; i < n; i += s) out[i] = (__bf16)in[i];
}

__global__ void zero_u32_kernel(unsigned int* __restrict__ p, long n32) {
  long i = (long)blockIdx.x * blockDim.x + threadIdx.x;
  const long s = (long)gridDim.x * blockDim.x;
  for (; i < n32; i += s) p[i] = 0u;
}

// ------------- host entry -------------
extern "C" void kernel_launch(void* const* d_in, const int* in_sizes, int n_in,
                              void* d_out, int out_size, void* d_ws,
                              size_t ws_size, hipStream_t stream) {
  const float* enc = (const float*)d_in[0];  // [8192, 2048]
  const float* wih = (const float*)d_in[1];  // [6144, 2048]
  const float* whh = (const float*)d_in[2];  // [6144, 2048]
  const float* bih = (const float*)d_in[3];  // [6144]
  const float* bhh = (const float*)d_in[4];  // [6144]
  float*       out = (float*)d_out;          // [8192, 2048]

  char* ws = (char*)d_ws;
  __bf16* enc_bf = (__bf16*)ws;  ws += (size_t)BB * KK * 2;  //  32 MB
  __bf16* wih_bf = (__bf16*)ws;  ws += (size_t)H3 * KK * 2;  //  24 MB
  __bf16* whh_bf = (__bf16*)ws;  ws += (size_t)H3 * KK * 2;  //  24 MB
  __bf16* gi_bf  = (__bf16*)ws;  ws += (size_t)BB * H3 * 2;  //  96 MB
  __bf16* hA     = (__bf16*)ws;  ws += (size_t)BB * HH * 2;  //  32 MB
  __bf16* hB     = (__bf16*)ws;                              //  32 MB

  // 1) precision conversion (one-time)
  cvt_f32_bf16_kernel<<<2048, 256, 0, stream>>>(enc, enc_bf, (long)BB * KK);
  cvt_f32_bf16_kernel<<<2048, 256, 0, stream>>>(wih, wih_bf, (long)H3 * KK);
  cvt_f32_bf16_kernel<<<2048, 256, 0, stream>>>(whh, whh_bf, (long)H3 * KK);
  zero_u32_kernel<<<2048, 256, 0, stream>>>((unsigned int*)hA,
                                            (long)BB * HH / 2);

  // 2) gi = enc @ w_ihT + b_ih
  dim3 grid(BB / BM, HH / BN);  // 64 x 32 blocks
  gi_gemm_kernel<<<grid, 256, 0, stream>>>(enc_bf, wih_bf, bih, gi_bf);

  // 3) 10 fused recurrence steps (h ping-pongs in bf16)
  for (int it = 0; it < 10; ++it) {
    const __bf16* hin  = (it & 1) ? hB : hA;
    __bf16*       hout = (it & 1) ? hA : hB;
    gru_step_kernel<<<grid, 256, 0, stream>>>(hin, whh_bf, gi_bf, bhh, hout,
                                              (it == 9) ? out : nullptr);
  }
}
